// GeometricAttention_26826365731148
// MI455X (gfx1250) — matrix operand order repaired and checked by reading, not compiled, not run on hardware
//
#include <hip/hip_runtime.h>
#include <math.h>

#define DEV_INLINE __device__ __forceinline__

typedef float v8f __attribute__((ext_vector_type(8)));
typedef unsigned int v4u __attribute__((ext_vector_type(4)));
typedef int v4i __attribute__((ext_vector_type(4)));
typedef int v8i __attribute__((ext_vector_type(8)));
typedef __bf16 v16bf __attribute__((ext_vector_type(16)));
typedef __bf16 v8bf __attribute__((ext_vector_type(8)));
typedef __bf16 v4bf __attribute__((ext_vector_type(4)));

// Problem constants (match reference)
constexpr int Bc = 4, Sc = 1024, DMc = 1024, Hc = 16, DKc = 64, FDc = 16;

DEV_INLINE v8f vzero8() {
  v8f z;
#pragma unroll
  for (int i = 0; i < 8; ++i) z[i] = 0.0f;
  return z;
}

// bf16 WMMA: D(16x16 f32) = A(16x32 bf16) * B(32x16 bf16) + C
DEV_INLINE v8f wmma_bf16(v16bf a, v16bf b, v8f c) {
  return __builtin_amdgcn_wmma_f32_16x16x32_bf16(false, a, false, b, (short)0,
                                                 c, false, false);
}

DEV_INLINE v16bf cat8(v8bf lo, v8bf hi) {
  return __builtin_shufflevector(lo, hi, 0, 1, 2, 3, 4, 5, 6, 7, 8, 9, 10, 11,
                                 12, 13, 14, 15);
}
DEV_INLINE v4bf cvt4(const float4 f) {
  v4bf r;
  r[0] = (__bf16)f.x;
  r[1] = (__bf16)f.y;
  r[2] = (__bf16)f.z;
  r[3] = (__bf16)f.w;
  return r;
}

// ---------------------------------------------------------------------------
// Tensor Data Mover: 2D tile (tile_d1 rows x tile_d0 f32 elems) -> LDS with
// row padding (pad_ilog: 3 -> pad every 16 DWORDs, 5 -> every 64; pad_m1=3 ->
// +4 DWORDs), giving the bank-conflict-free 68/20-float LDS strides natively.
// ---------------------------------------------------------------------------
#if __has_builtin(__builtin_amdgcn_tensor_load_to_lds)
#define HAVE_TDM 1
DEV_INLINE void tdm_load_2d(unsigned lds_addr, const void* gptr,
                            unsigned tile_d0, unsigned tile_d1,
                            unsigned row_stride, unsigned pad_ilog,
                            unsigned pad_m1) {
  const unsigned long long ga = (unsigned long long)gptr;
  v4u g0;
  g0[0] = 1u;                                  // count=1 (valid descriptor)
  g0[1] = lds_addr;                            // lds_addr[31:0]
  g0[2] = (unsigned)(ga & 0xFFFFFFFFu);        // global_addr[31:0]
  g0[3] = (unsigned)((ga >> 32) & 0x1FFFFFFu)  // global_addr[56:32]
          | (2u << 30);                        // type = 2 (image)
  v8i g1;
  g1[0] = (int)((2u << 16)                     // data_size = 4B
                | (1u << 20)                   // pad_enable
                | (pad_ilog << 22)             // pad_interval
                | (pad_m1 << 25));             // pad_amount
  g1[1] = (int)(tile_d0 << 16);                // tensor_dim0[15:0]
  g1[2] = (int)(tile_d1 << 16);                // tensor_dim1[15:0]
  g1[3] = (int)(tile_d0 << 16);                // tile_dim0
  g1[4] = (int)tile_d1;                        // tile_dim1 (tile_dim2=0)
  g1[5] = (int)row_stride;                     // tensor_dim0_stride[31:0]
  g1[6] = 0;
  g1[7] = 0;
  const v4i z = {0, 0, 0, 0};
#if __clang_major__ >= 23
  const v8i z8 = {0, 0, 0, 0, 0, 0, 0, 0};
  __builtin_amdgcn_tensor_load_to_lds(g0, g1, z, z, z8, 0);
#else
  __builtin_amdgcn_tensor_load_to_lds(g0, g1, z, z, 0);
#endif
}
DEV_INLINE unsigned lds_off(const void* p) {
  // LDS flat aperture: hardware LDS address is addr[31:0] of the flat pointer.
  return (unsigned)(unsigned long long)p;
}
#else
#define HAVE_TDM 0
#endif

// ---------------------------------------------------------------------------
// Kernel 0: T_avg = mean over (8,8) of T_field(8,8,16,16); T_proj = Wf @ T_avg
// ---------------------------------------------------------------------------
__global__ __launch_bounds__(256) void tproj_kernel(
    const float* __restrict__ T_field, const float* __restrict__ Wf,
    float* __restrict__ T_proj) {
  __shared__ float tavg[FDc * FDc];
  const int t = threadIdx.x;
  float acc = 0.0f;
#pragma unroll 4
  for (int ab = 0; ab < 64; ++ab) acc += T_field[ab * 256 + t];
  tavg[t] = acc * (1.0f / 64.0f);
  __syncthreads();
  for (int r = t; r < DKc * DKc; r += blockDim.x) {
    const float* w = Wf + (size_t)r * 256;
    float s = 0.0f;
#pragma unroll 8
    for (int j = 0; j < 256; ++j) s += w[j] * tavg[j];
    T_proj[r] = s;
  }
}

// ---------------------------------------------------------------------------
// Kernel 1: C[m,n] = sum_k A[m,k]*W[n,k] + bias[n]  via bf16 WMMA (f32 acc).
// Block (8 waves) computes 64x128 C tile; K chunked (KC=64) through double-
// buffered bf16 LDS tiles. All 10 fragments of a chunk are preloaded into
// registers before the 8 WMMAs so ds_load latency overlaps the matrix pipe.
// ---------------------------------------------------------------------------
__global__ __launch_bounds__(256) void wmma_proj_kernel(
    const float* __restrict__ A, const float* __restrict__ W,
    const float* __restrict__ bias, float* __restrict__ C, int M, int K, int N,
    int split_heads) {
  __shared__ __bf16 At[2][64][72];  // stride 72 bf16 = 36 DWORDs: rows distinct
  __shared__ __bf16 Wt[2][128][72];

  const int tid = threadIdx.x;
  const int lane = tid & 31;
  const int wave = tid >> 5;
  const int l15 = lane & 15;
  const int halfk = lane >> 4;
  const int wm = wave >> 1;
  const int wn = wave & 1;

  const int colBlocks = N >> 7;
  const int rowBlk = blockIdx.x / colBlocks;
  const int colBlk = blockIdx.x % colBlocks;
  const int m0 = rowBlk * 64;
  const int n0 = colBlk * 128;

  constexpr int KC = 64;
  const int nCh = K / KC;

  // Stage chunk 0 into buffer 0 (fp32 -> bf16 fused into the LDS store).
#pragma unroll
  for (int i = 0; i < 4; ++i) {
    const int idx = tid + i * 256;
    const int row = idx >> 4, col = (idx & 15) << 2;
    *(v4bf*)&At[0][row][col] =
        cvt4(*(const float4*)(A + (size_t)(m0 + row) * K + col));
  }
#pragma unroll
  for (int i = 0; i < 8; ++i) {
    const int idx = tid + i * 256;
    const int row = idx >> 4, col = (idx & 15) << 2;
    *(v4bf*)&Wt[0][row][col] =
        cvt4(*(const float4*)(W + (size_t)(n0 + row) * K + col));
  }
  __syncthreads();

  v8f acc[4];
#pragma unroll
  for (int t = 0; t < 4; ++t) acc[t] = vzero8();

  const int arow = wm * 16 + l15;
  for (int c = 0; c < nCh; ++c) {
    const int buf = c & 1;
    const bool pre = (c + 1 < nCh);
    float4 ra[4], rw[8];
    if (pre) {
      const int kc = (c + 1) * KC;
#pragma unroll
      for (int i = 0; i < 4; ++i) {
        const int idx = tid + i * 256;
        ra[i] = *(const float4*)(A + (size_t)(m0 + (idx >> 4)) * K + kc +
                                 ((idx & 15) << 2));
      }
#pragma unroll
      for (int i = 0; i < 8; ++i) {
        const int idx = tid + i * 256;
        rw[i] = *(const float4*)(W + (size_t)(n0 + (idx >> 4)) * K + kc +
                                 ((idx & 15) << 2));
      }
    }
    // Preload all fragments of this chunk, then fire the 8 WMMAs.
    v16bf afr[2], bfr[2][4];
#pragma unroll
    for (int ks = 0; ks < 2; ++ks) {
      const int kb = ks * 32;
      // A lane layout (16-bit A 16x32): K in {16h..16h+15}
      afr[ks] = cat8(*(const v8bf*)&At[buf][arow][kb + halfk * 16],
                     *(const v8bf*)&At[buf][arow][kb + halfk * 16 + 8]);
#pragma unroll
      for (int t = 0; t < 4; ++t) {
        const int wrow = wn * 64 + t * 16 + l15;
        // B lane layout (32x16): K in {16h..16h+15}
        bfr[ks][t] = cat8(*(const v8bf*)&Wt[buf][wrow][kb + halfk * 16],
                          *(const v8bf*)&Wt[buf][wrow][kb + halfk * 16 + 8]);
      }
    }
#pragma unroll
    for (int ks = 0; ks < 2; ++ks)
#pragma unroll
      for (int t = 0; t < 4; ++t)
        acc[t] = wmma_bf16(afr[ks], bfr[ks][t], acc[t]);

    if (pre) {
      const int nb = buf ^ 1;
#pragma unroll
      for (int i = 0; i < 4; ++i) {
        const int idx = tid + i * 256;
        *(v4bf*)&At[nb][idx >> 4][(idx & 15) << 2] = cvt4(ra[i]);
      }
#pragma unroll
      for (int i = 0; i < 8; ++i) {
        const int idx = tid + i * 256;
        *(v4bf*)&Wt[nb][idx >> 4][(idx & 15) << 2] = cvt4(rw[i]);
      }
    }
    __syncthreads();
  }

  const int wm0 = m0 + wm * 16;
  const int wn0 = n0 + wn * 64;
#pragma unroll
  for (int t = 0; t < 4; ++t) {
    const int n = wn0 + t * 16 + l15;
    const float bv = bias[n];
#pragma unroll
    for (int r = 0; r < 8; ++r) {
      const int m = wm0 + r + 8 * halfk;
      const float val = acc[t][r] + bv;
      if (split_heads) {
        const int bb = m / Sc, ss = m % Sc, hh = n / DKc, dd = n % DKc;
        C[(((size_t)bb * Hc + hh) * Sc + ss) * DKc + dd] = val;
      } else {
        C[(size_t)m * N + n] = val;
      }
    }
  }
}

// ---------------------------------------------------------------------------
// Kernel 2: psi = exp_phase(x) per (b,h,s).
// ---------------------------------------------------------------------------
__global__ __launch_bounds__(256) void psi_phase_kernel(
    const float* __restrict__ X, const float* __restrict__ Wp,
    const float* __restrict__ bp, float* __restrict__ psi, int bhsd_layout) {
  const int idx = blockIdx.x * blockDim.x + threadIdx.x;
  if (idx >= Bc * Hc * Sc) return;
  const int s = idx % Sc;
  const int h = (idx / Sc) % Hc;
  const int b = idx / (Sc * Hc);
  const float* x = bhsd_layout ? (X + (size_t)idx * DKc)
                               : (X + ((size_t)b * Sc + s) * DMc + h * DKc);
  float xv[DKc];
#pragma unroll
  for (int d = 0; d < DKc; ++d) xv[d] = x[d];

  float* o = psi + (size_t)idx * 16;
#pragma unroll
  for (int j = 0; j < 8; ++j) {
    float ga = bp[j], gb = bp[8 + j], gt = bp[16 + j];
    const float* wa = Wp + (size_t)j * DKc;
    const float* wb = Wp + (size_t)(8 + j) * DKc;
    const float* wt = Wp + (size_t)(16 + j) * DKc;
#pragma unroll 8
    for (int d = 0; d < DKc; ++d) {
      ga += wa[d] * xv[d];
      gb += wb[d] * xv[d];
      gt += wt[d] * xv[d];
    }
    gt = fminf(8.0f, fmaxf(-8.0f, gt));
    o[j] = ga * __expf(gt) + gb * __cosf(gt);
    o[8 + j] = gb * __sinf(gt);
  }
}

// ---------------------------------------------------------------------------
// Kernel 3: fused geometric flash attention, bf16 WMMA core.
// Per tile: TDM lands f32 K/V/psi_k tiles (double-buffered); a cooperative
// 128-thread pass converts them ONCE to bf16 LDS buffers (V transposed so
// B-fragments are contiguous); waves then feed WMMAs with pure ds_load_b128.
// Row norms / softmax / accumulators stay f32.
// ---------------------------------------------------------------------------
__global__ __launch_bounds__(128) void flash_geo_attn_kernel(
    const float* __restrict__ Qh, const float* __restrict__ Kin,
    const float* __restrict__ Vin, const float* __restrict__ psiQ,
    const float* __restrict__ psiK, const float* __restrict__ Tproj,
    const float* __restrict__ gw, const float* __restrict__ temperature,
    const float* __restrict__ field_alpha, float* __restrict__ Out1) {
  __shared__ float Ktile[2][16][68];  // TDM pads 64-DWORD rows by +4 -> 68
  __shared__ float Vtile[2][16][68];
  __shared__ float psiKs[2][16][20];  // TDM pads 16-DWORD rows by +4 -> 20
  __shared__ __bf16 Kbf[16][72];      // bf16 K tile, row-major
  __shared__ __bf16 VbfT[64][24];     // bf16 V tile, TRANSPOSED [n][k]
  __shared__ __bf16 psiKbf[16][24];   // bf16 psi_k tile, row-major
  __shared__ float Ps[4][16][20];     // per-wave P buffer (C->A relayout)

  const int tid = threadIdx.x;
  const int lane = tid & 31;
  const int wave = tid >> 5;
  const int l15 = lane & 15;
  const int halfk = lane >> 4;

  const int qblocks = Sc / 64;
  const int bh = blockIdx.x / qblocks;
  const int qblk = blockIdx.x % qblocks;
  const int b = bh / Hc;
  const int h = bh % Hc;
  const int qrow0 = qblk * 64 + wave * 16;

  const float gw0 = gw[0], gw1 = gw[1], gw2 = gw[2], gw3 = gw[3];
  const float invT = 1.0f / temperature[0];
  const float alpha = field_alpha[0];

  // --- Q fragments, bf16 (A 16x32 layout: K {16h..16h+15})
  const float* Qbase = Qh + (((size_t)bh) * Sc + qrow0) * DKc;
  v16bf qf[2];
  float qn2p = 0.0f;
#pragma unroll
  for (int j = 0; j < 2; ++j) {
    const float* p0 = Qbase + (size_t)l15 * DKc + 32 * j + halfk * 16;
    const float* p1 = Qbase + (size_t)l15 * DKc + 32 * j + halfk * 16 + 8;
#pragma unroll
    for (int e = 0; e < 8; ++e) {
      qf[j][e] = (__bf16)p0[e];
      qf[j][8 + e] = (__bf16)p1[e];
      qn2p += p0[e] * p0[e] + p1[e] * p1[e];
    }
  }
  qn2p += __shfl_xor(qn2p, 16, 32);
  float qn2v[8];
#pragma unroll
  for (int r = 0; r < 8; ++r) qn2v[r] = __shfl(qn2p, r + 8 * halfk, 32);

  // --- psi_q fragment, K=16 zero-padded to 32 (A K>=16 slots are zero)
  const float* PQbase = psiQ + (((size_t)bh) * Sc + qrow0) * 16;
  v16bf pqf;
  {
    const float* pp = PQbase + (size_t)l15 * 16 + halfk * 8;
#pragma unroll
    for (int e = 0; e < 8; ++e) {
      pqf[e] = (__bf16)pp[e];
      pqf[8 + e] = (__bf16)0.0f;
    }
  }

  v8f acc[4];
#pragma unroll
  for (int t = 0; t < 4; ++t) acc[t] = vzero8();
  float mrow[8], lrow[8];
#pragma unroll
  for (int r = 0; r < 8; ++r) {
    mrow[r] = -1.0e30f;
    lrow[r] = 0.0f;
  }

  const float* KbaseBH = Kin + ((size_t)b * Sc) * DMc + h * DKc;
  const float* VbaseBH = Vin + ((size_t)b * Sc) * DMc + h * DKc;
  const float* PKbase = psiK + ((size_t)bh) * Sc * 16;

#if HAVE_TDM
  if (wave == 0) {  // prime: tile 0 -> buffer 0
    tdm_load_2d(lds_off(&Ktile[0][0][0]), KbaseBH, 64, 16, DMc, 5, 3);
    tdm_load_2d(lds_off(&Vtile[0][0][0]), VbaseBH, 64, 16, DMc, 5, 3);
    tdm_load_2d(lds_off(&psiKs[0][0][0]), PKbase, 16, 16, 16, 3, 3);
  }
#endif

  const int nTiles = Sc / 16;
  for (int kt = 0; kt < nTiles; ++kt) {
    const int cur = kt & 1;
    const int k0 = kt * 16;
#if HAVE_TDM
    if (wave == 0) {
      if (kt + 1 < nTiles) {
        const int nk0 = k0 + 16;
        const int nb = cur ^ 1;
        tdm_load_2d(lds_off(&Ktile[nb][0][0]), KbaseBH + (size_t)nk0 * DMc, 64,
                    16, DMc, 5, 3);
        tdm_load_2d(lds_off(&Vtile[nb][0][0]), VbaseBH + (size_t)nk0 * DMc, 64,
                    16, DMc, 5, 3);
        tdm_load_2d(lds_off(&psiKs[nb][0][0]), PKbase + (size_t)nk0 * 16, 16,
                    16, 16, 3, 3);
        __builtin_amdgcn_s_wait_tensorcnt(3);  // in-order: current tile landed
      } else {
        __builtin_amdgcn_s_wait_tensorcnt(0);
      }
    }
#else
#pragma unroll
    for (int i = 0; i < 2; ++i) {
      const int idx = tid + i * 128;
      const int row = idx >> 4, col = (idx & 15) << 2;
      *(float4*)&Ktile[cur][row][col] =
          *(const float4*)(KbaseBH + ((size_t)(k0 + row)) * DMc + col);
      *(float4*)&Vtile[cur][row][col] =
          *(const float4*)(VbaseBH + ((size_t)(k0 + row)) * DMc + col);
    }
    if (tid < 64) {
      const int row = tid >> 2, col = (tid & 3) << 2;
      *(float4*)&psiKs[cur][row][col] =
          *(const float4*)(PKbase + ((size_t)(k0 + row)) * 16 + col);
    }
#endif
    __syncthreads();  // f32 tile `cur` visible to all waves

    // --- cooperative f32 -> bf16 conversion pass (once per block, not wave)
    {
      // K tile: 16 rows x 64; 8 threads/row, 8 elems each
      const int krow = tid >> 3, kcol = (tid & 7) << 3;
      v8bf kk;
#pragma unroll
      for (int e = 0; e < 8; ++e)
        kk[e] = (__bf16)Ktile[cur][krow][kcol + e];
      *(v8bf*)&Kbf[krow][kcol] = kk;
      // V tile transposed: 64 cols (n); 2 threads/col, 8 k each
      const int vn = tid >> 1, vk = (tid & 1) << 3;
      v8bf vv;
#pragma unroll
      for (int e = 0; e < 8; ++e)
        vv[e] = (__bf16)Vtile[cur][vk + e][vn];
      *(v8bf*)&VbfT[vn][vk] = vv;
      // psi_k tile: 16 rows x 16; 2 threads/row, 8 elems each
      if (tid < 32) {
        const int pr = tid >> 1, pc = (tid & 1) << 3;
        v8bf pp;
#pragma unroll
        for (int e = 0; e < 8; ++e)
          pp[e] = (__bf16)psiKs[cur][pr][pc + e];
        *(v8bf*)&psiKbf[pr][pc] = pp;
      }
    }
    __syncthreads();  // bf16 buffers visible

    // |k|^2 for key n = l15 (from f32 tile)
    float kp = 0.0f;
#pragma unroll 8
    for (int dd = 0; dd < 32; ++dd) {
      const float tv = Ktile[cur][l15][halfk * 32 + dd];
      kp += tv * tv;
    }
    const float kn2n = kp + __shfl_xor(kp, 16, 32);

    // raw = Q(16x64) @ K^T : 2 bf16 WMMA, fragments straight from LDS
    const v16bf kb0 = cat8(*(const v8bf*)&Kbf[l15][halfk * 16],
                           *(const v8bf*)&Kbf[l15][halfk * 16 + 8]);
    const v16bf kb1 = cat8(*(const v8bf*)&Kbf[l15][32 + halfk * 16],
                           *(const v8bf*)&Kbf[l15][32 + halfk * 16 + 8]);
    v8f raws = vzero8();
    raws = wmma_bf16(qf[0], kb0, raws);
    raws = wmma_bf16(qf[1], kb1, raws);
    // spinor (K=16 zero-padded; B elements 8..15 are don't-care vs zero A)
    const v16bf pkb = cat8(*(const v8bf*)&psiKbf[l15][halfk * 8],
                           *(const v8bf*)&psiKbf[l15][8]);
    v8f sp = vzero8();
    sp = wmma_bf16(pqf, pkb, sp);

    // scores + online softmax (C layout: lane = col n, VGPR r = row r+8h)
#pragma unroll
    for (int r = 0; r < 8; ++r) {
      const float rv = raws[r];
      const float sc = (gw0 * ((qn2v[r] * kn2n - rv * rv) * (1.0f / 4096.0f)) +
                        gw1 * (rv * rv * (1.0f / 64.0f)) +
                        gw2 * (sp[r] * 0.0625f) + gw3 * (rv * 0.125f)) *
                       invT;
      float rmax = sc;
      rmax = fmaxf(rmax, __shfl_xor(rmax, 1, 32));
      rmax = fmaxf(rmax, __shfl_xor(rmax, 2, 32));
      rmax = fmaxf(rmax, __shfl_xor(rmax, 4, 32));
      rmax = fmaxf(rmax, __shfl_xor(rmax, 8, 32));
      const float mnew = fmaxf(mrow[r], rmax);
      const float pv = __expf(sc - mnew);
      const float corr = __expf(mrow[r] - mnew);
      float rsum = pv;
      rsum += __shfl_xor(rsum, 1, 32);
      rsum += __shfl_xor(rsum, 2, 32);
      rsum += __shfl_xor(rsum, 4, 32);
      rsum += __shfl_xor(rsum, 8, 32);
      lrow[r] = lrow[r] * corr + rsum;
      mrow[r] = mnew;
#pragma unroll
      for (int t = 0; t < 4; ++t) acc[t][r] *= corr;
      Ps[wave][r + 8 * halfk][l15] = pv;
    }

    // acc += P(16x16) @ V(16x64): K=16 zero-padded A; B frags are contiguous
    // rows of the transposed bf16 V tile. Same-wave DS ops in-order.
    v16bf paf;
    {
      const float* pr = &Ps[wave][l15][halfk * 8];
#pragma unroll
      for (int e = 0; e < 8; ++e) {
        paf[e] = (__bf16)pr[e];
        paf[8 + e] = (__bf16)0.0f;
      }
    }
    v16bf vfr[4];
#pragma unroll
    for (int t = 0; t < 4; ++t)
      vfr[t] = cat8(*(const v8bf*)&VbfT[t * 16 + l15][halfk * 8],
                    *(const v8bf*)&VbfT[t * 16 + l15][8]);
#pragma unroll
    for (int t = 0; t < 4; ++t) acc[t] = wmma_bf16(paf, vfr[t], acc[t]);

    __syncthreads();  // block done with tile `cur` + bf16 buffers
  }

  // normalize by softmax denominator
#pragma unroll
  for (int r = 0; r < 8; ++r) {
    const float inv = 1.0f / lrow[r];
#pragma unroll
    for (int t = 0; t < 4; ++t) acc[t][r] *= inv;
  }

  // field term: Vc = Kh_rows(16x64) @ Tproj^T, bf16 WMMA (B[k][n]=Tproj[n][k])
  v8f vc[4];
#pragma unroll
  for (int t = 0; t < 4; ++t) vc[t] = vzero8();
  const float* KQrow = Kin + ((size_t)b * Sc + qrow0) * DMc + h * DKc;
  v16bf kaf[2];
#pragma unroll
  for (int j = 0; j < 2; ++j) {
    const float* p0 = KQrow + (size_t)l15 * DMc + 32 * j + halfk * 16;
    const float* p1 = KQrow + (size_t)l15 * DMc + 32 * j + halfk * 16 + 8;
#pragma unroll
    for (int e = 0; e < 8; ++e) {
      kaf[j][e] = (__bf16)p0[e];
      kaf[j][8 + e] = (__bf16)p1[e];
    }
  }
#pragma unroll
  for (int j = 0; j < 2; ++j) {
#pragma unroll
    for (int t = 0; t < 4; ++t) {
      const int n = t * 16 + l15;
      const float* tp = Tproj + n * 64 + 32 * j + halfk * 16;
      v16bf tbf;
#pragma unroll
      for (int e = 0; e < 16; ++e) tbf[e] = (__bf16)tp[e];
      vc[t] = wmma_bf16(kaf[j], tbf, vc[t]);
    }
  }

  // store merged-head output [B, S, DM]
  float* Obase = Out1 + ((size_t)b * Sc + qrow0) * DMc + h * DKc;
#pragma unroll
  for (int t = 0; t < 4; ++t) {
#pragma unroll
    for (int r = 0; r < 8; ++r) {
      const int m = r + 8 * halfk;
      Obase[(size_t)m * DMc + t * 16 + l15] = acc[t][r] + alpha * vc[t][r];
    }
  }
}

// ---------------------------------------------------------------------------
extern "C" void kernel_launch(void* const* d_in, const int* in_sizes, int n_in,
                              void* d_out, int out_size, void* d_ws,
                              size_t ws_size, hipStream_t stream) {
  (void)in_sizes;
  (void)n_in;
  (void)out_size;
  (void)ws_size;
  const float* Q = (const float*)d_in[0];
  const float* Kin = (const float*)d_in[1];
  const float* Vin = (const float*)d_in[2];
  const float* T_field = (const float*)d_in[3];
  const float* Wq = (const float*)d_in[4];
  const float* bq = (const float*)d_in[5];
  const float* Wo = (const float*)d_in[6];
  const float* bo = (const float*)d_in[7];
  const float* Wp = (const float*)d_in[8];
  const float* bp = (const float*)d_in[9];
  const float* Wf = (const float*)d_in[10];
  const float* gwp = (const float*)d_in[11];
  const float* temp = (const float*)d_in[12];
  const float* alpha = (const float*)d_in[13];
  float* out = (float*)d_out;

  float* ws = (float*)d_ws;
  float* Qh = ws;                                   // B*H*S*DK = 4,194,304
  float* psiQ = Qh + (size_t)Bc * Hc * Sc * DKc;    // B*H*S*16 = 1,048,576
  float* psiK = psiQ + (size_t)Bc * Hc * Sc * 16;   // B*H*S*16 = 1,048,576
  float* Tproj = psiK + (size_t)Bc * Hc * Sc * 16;  // 64*64
  float* Out1 = Tproj + (size_t)DKc * DKc;          // B*S*DM   = 4,194,304

  tproj_kernel<<<1, 256, 0, stream>>>(T_field, Wf, Tproj);

  const int M = Bc * Sc, Kd = DMc, N = DMc;
  const int nBlocks = (M / 64) * (N / 128);  // 512 blocks, 8 waves each
  wmma_proj_kernel<<<nBlocks, 256, 0, stream>>>(Q, Wq, bq, Qh, M, Kd, N, 1);

  const int npos = Bc * Hc * Sc;
  psi_phase_kernel<<<npos / 256, 256, 0, stream>>>(Qh, Wp, bp, psiQ, 1);
  psi_phase_kernel<<<npos / 256, 256, 0, stream>>>(Kin, Wp, bp, psiK, 0);

  flash_geo_attn_kernel<<<Bc * Hc * (Sc / 64), 128, 0, stream>>>(
      Qh, Kin, Vin, psiQ, psiK, Tproj, gwp, temp, alpha, Out1);

  wmma_proj_kernel<<<nBlocks, 256, 0, stream>>>(Out1, Wo, bo, out, M, Kd, N, 0);
}
